// FastMultiHeadAttention_64854006169987
// MI455X (gfx1250) — matrix operand order repaired and checked by reading, not compiled, not run on hardware
//
#include <hip/hip_runtime.h>
#include <hip/hip_bf16.h>

#define D_EMBED 1024
#define N_HEADS 16
#define HD      64
#define SEQ     2048
#define BATCH   2
#define MROWS   (BATCH*SEQ)   /* 4096 */
#define NQKV    (3*D_EMBED)   /* 3072 */
#define RANK    8
#define LSCALE  2.0f          /* alpha/r = 16/8 */

typedef __bf16 bf16_t;
typedef __attribute__((ext_vector_type(16))) __bf16 v16bf;
typedef __attribute__((ext_vector_type(8)))  __bf16 v8bf;
typedef __attribute__((ext_vector_type(8)))  float  v8f;
typedef __attribute__((ext_vector_type(4)))  unsigned int v4u;
typedef __attribute__((ext_vector_type(4)))  int v4i;
typedef __attribute__((ext_vector_type(8)))  int v8i;

union Frag { v16bf v; v8bf h[2]; };

static __device__ __forceinline__ __bf16 f2bf(float x) { return (__bf16)x; }

// -------------------------------------------------------------- TDM 2D tile load
// Builds a Tensor DMA Descriptor (D#) per CDNA5 ISA ch.8 and issues
// tensor_load_to_lds. dims/tiles in elements (data_size = 2 bytes, bf16).
static __device__ __forceinline__ void tdm_load_2d(unsigned lds_off,
                                                   unsigned long long gaddr,
                                                   unsigned tile0, unsigned tile1,
                                                   unsigned dim0, unsigned dim1,
                                                   unsigned long long stride0) {
  v4u g0;
  g0.x = 1u;                                    // count=1, user descriptor
  g0.y = lds_off;                               // lds_addr [63:32]
  g0.z = (unsigned)(gaddr & 0xffffffffu);       // global_addr [95:64]
  g0.w = (unsigned)((gaddr >> 32) & 0x1ffffffu) // global_addr [120:96]
         | (2u << 30);                          // type=2 ("image") [127:126]
  v8i g1;
  g1[0] = (int)(1u << 16);                      // data_size=1 (2 bytes), wg_mask=0
  g1[1] = (int)((dim0 & 0xffffu) << 16);        // tensor_dim0 lo16 @ [63:48]
  g1[2] = (int)(((dim0 >> 16) & 0xffffu) | ((dim1 & 0xffffu) << 16)); // dim0 hi / dim1 lo
  g1[3] = (int)(((dim1 >> 16) & 0xffffu) | ((tile0 & 0xffffu) << 16)); // dim1 hi / tile_dim0
  g1[4] = (int)(tile1 & 0xffffu);               // tile_dim1 (tile_dim2=0)
  g1[5] = (int)(unsigned)(stride0 & 0xffffffffu);      // dim0_stride lo32
  g1[6] = (int)(unsigned)((stride0 >> 32) & 0xffffu);  // dim0_stride hi16
  g1[7] = 0;
  v4i z4 = {};
#if __clang_major__ >= 23
  v8i z8 = {};
  __builtin_amdgcn_tensor_load_to_lds(g0, g1, z4, z4, z8, 0);
#else
  __builtin_amdgcn_tensor_load_to_lds(g0, g1, z4, z4, 0);
#endif
}

static __device__ __forceinline__ unsigned lds_offset_of(const void* p) {
  return (unsigned)(unsigned long long)(uintptr_t)p;  // low 32 bits = LDS offset
}

// ---------------------------------------------------------------- fp32 -> bf16
__global__ __launch_bounds__(256) void k_f32_to_bf16(const float* __restrict__ in,
                                                     bf16_t* __restrict__ out, int n) {
  int i = blockIdx.x * blockDim.x + threadIdx.x;
  if (i < n) out[i] = f2bf(in[i]);
}

// ------------------------------------------------- LoRA intermediate t = X @ A^T
__global__ __launch_bounds__(256) void k_lora_t_f32(const float* __restrict__ X,
                                                    const float* __restrict__ A,
                                                    float* __restrict__ T) {
  int t = blockIdx.x * blockDim.x + threadIdx.x;   // MROWS*RANK threads
  int m = t >> 3, j = t & 7;
  const float* xr = X + (size_t)m * D_EMBED;
  const float* ar = A + (size_t)j * D_EMBED;
  float s = 0.f;
#pragma unroll 8
  for (int k = 0; k < D_EMBED; ++k) s += xr[k] * ar[k];
  T[t] = s;
}

__global__ __launch_bounds__(256) void k_lora_t_bf16(const bf16_t* __restrict__ X,
                                                     const float* __restrict__ A,
                                                     float* __restrict__ T) {
  int t = blockIdx.x * blockDim.x + threadIdx.x;
  int m = t >> 3, j = t & 7;
  const bf16_t* xr = X + (size_t)m * D_EMBED;
  const float*  ar = A + (size_t)j * D_EMBED;
  float s = 0.f;
#pragma unroll 8
  for (int k = 0; k < D_EMBED; ++k) s += (float)xr[k] * ar[k];
  T[t] = s;
}

// --------------------------------------------------- WMMA GEMM, TDM double-buffer
// C[M,N] = A[M,K] @ W[N,K]^T + LSCALE * TL[M,r] @ BL[N,r]^T  (+ bias, MODE 1)
template <int MODE>   // 0: bf16 out (QKV), 1: fp32 out + bias (proj)
__global__ __launch_bounds__(256) void k_gemm_wmma(
    const bf16_t* __restrict__ A, const bf16_t* __restrict__ W,
    const float* __restrict__ TL, const float* __restrict__ BL,
    const float* __restrict__ bias,
    bf16_t* __restrict__ outb, float* __restrict__ outf,
    int M, int N, int K) {
  __shared__ alignas(16) bf16_t As[2][128][32];
  __shared__ alignas(16) bf16_t Ws[2][128][32];
  const int tid = threadIdx.x, wave = tid >> 5, lane = tid & 31;
  const int bm = blockIdx.y * 128, bn = blockIdx.x * 128;
  const int wm = (wave >> 2) * 64, wn = (wave & 3) * 32;  // 2x4 waves: 64x32 each
  const int lr  = lane & 15;
  const int kA  = (lane >> 4) * 16;   // A-fragment K base (16 contiguous)
  const int kB  = (lane >> 4) * 16;   // B-fragment K base (16 contiguous)
  const int hi8 = (lane >> 4) * 8;
  v8f acc[4][2] = {};

  // Prologue: wave 0 kicks off the first pair of tile DMAs.
  if (wave == 0) {
    tdm_load_2d(lds_offset_of(&As[0][0][0]),
                (unsigned long long)(uintptr_t)(A + (size_t)bm * K),
                32, 128, 32, 128, (unsigned long long)K);
    tdm_load_2d(lds_offset_of(&Ws[0][0][0]),
                (unsigned long long)(uintptr_t)(W + (size_t)bn * K),
                32, 128, 32, 128, (unsigned long long)K);
  }

  int buf = 0;
  for (int k0 = 0; k0 < K; k0 += 32) {
    if (wave == 0) __builtin_amdgcn_s_wait_tensorcnt(0);
    __syncthreads();                       // current buffer ready for all waves
    if (wave == 0 && k0 + 32 < K) {        // overlap next DMA with compute
      tdm_load_2d(lds_offset_of(&As[buf ^ 1][0][0]),
                  (unsigned long long)(uintptr_t)(A + (size_t)bm * K + k0 + 32),
                  32, 128, 32, 128, (unsigned long long)K);
      tdm_load_2d(lds_offset_of(&Ws[buf ^ 1][0][0]),
                  (unsigned long long)(uintptr_t)(W + (size_t)bn * K + k0 + 32),
                  32, 128, 32, 128, (unsigned long long)K);
    }
    Frag bfr[2];
#pragma unroll
    for (int ni = 0; ni < 2; ++ni) {
      bfr[ni].h[0] = *(const v8bf*)&Ws[buf][wn + ni * 16 + lr][kB];
      bfr[ni].h[1] = *(const v8bf*)&Ws[buf][wn + ni * 16 + lr][kB + 8];
    }
#pragma unroll
    for (int mi = 0; mi < 4; ++mi) {
      Frag a;
      a.h[0] = *(const v8bf*)&As[buf][wm + mi * 16 + lr][kA];
      a.h[1] = *(const v8bf*)&As[buf][wm + mi * 16 + lr][kA + 8];
#pragma unroll
      for (int ni = 0; ni < 2; ++ni)
        acc[mi][ni] = __builtin_amdgcn_wmma_f32_16x16x32_bf16(
            false, a.v, false, bfr[ni].v, (short)0, acc[mi][ni], false, false);
    }
    buf ^= 1;
  }

  // Epilogue: rank-8 LoRA (+bias) fused per element.
  float blr[2][RANK];
#pragma unroll
  for (int ni = 0; ni < 2; ++ni) {
    int n = bn + wn + ni * 16 + lr;
#pragma unroll
    for (int j = 0; j < RANK; ++j) blr[ni][j] = BL[(size_t)n * RANK + j];
  }
#pragma unroll
  for (int mi = 0; mi < 4; ++mi) {
#pragma unroll
    for (int r = 0; r < 8; ++r) {
      int m = bm + wm + mi * 16 + r + hi8;
      float tl[RANK];
#pragma unroll
      for (int j = 0; j < RANK; ++j) tl[j] = TL[(size_t)m * RANK + j];
#pragma unroll
      for (int ni = 0; ni < 2; ++ni) {
        int n = bn + wn + ni * 16 + lr;
        float v = acc[mi][ni][r];
        float ls = 0.f;
#pragma unroll
        for (int j = 0; j < RANK; ++j) ls += tl[j] * blr[ni][j];
        v += LSCALE * ls;
        if (MODE == 1) { v += bias[n]; outf[(size_t)m * N + n] = v; }
        else           { outb[(size_t)m * N + n] = f2bf(v); }
      }
    }
  }
}

// --------------------------------------------------- RoPE + split to (B,H,S,hd)
__global__ __launch_bounds__(256) void k_rope_split(const bf16_t* __restrict__ qkv,
                                                    bf16_t* __restrict__ Qo,
                                                    bf16_t* __restrict__ Ko,
                                                    bf16_t* __restrict__ Vo) {
  int t = blockIdx.x * blockDim.x + threadIdx.x;   // B*H*S*32 threads
  int d = t & 31;
  int s = (t >> 5) & (SEQ - 1);
  int h = (t >> 16) & (N_HEADS - 1);
  int b = t >> 20;
  size_t row = (size_t)(b * SEQ + s) * NQKV;
  int c0 = h * HD + d;
  float q1 = (float)qkv[row + c0],               q2 = (float)qkv[row + c0 + 32];
  float k1 = (float)qkv[row + D_EMBED + c0],     k2 = (float)qkv[row + D_EMBED + c0 + 32];
  float v1 = (float)qkv[row + 2 * D_EMBED + c0], v2 = (float)qkv[row + 2 * D_EMBED + c0 + 32];
  // inv_freq = 10000^(-d/32) ; ln(10000)/32 = 0.28782313662
  float ang = (float)s * __expf(-0.28782313662425572f * (float)d);
  float sn, cs;
  __sincosf(ang, &sn, &cs);
  size_t orow = ((size_t)(b * N_HEADS + h) * SEQ + s) * HD;
  Qo[orow + d]      = f2bf(q1 * cs - q2 * sn);
  Qo[orow + d + 32] = f2bf(q2 * cs + q1 * sn);
  Ko[orow + d]      = f2bf(k1 * cs - k2 * sn);
  Ko[orow + d + 32] = f2bf(k2 * cs + k1 * sn);
  Vo[orow + d]      = f2bf(v1);
  Vo[orow + d + 32] = f2bf(v2);
}

// ----------------------------------------------- flash attention, 1 q-tile/wave
__global__ __launch_bounds__(256) void k_flash_attn(const bf16_t* __restrict__ Qm,
                                                    const bf16_t* __restrict__ Km,
                                                    const bf16_t* __restrict__ Vm,
                                                    bf16_t* __restrict__ ctx) {
  __shared__ alignas(16) bf16_t pbuf[8][16][32];  // P tile per wave
  __shared__ alignas(16) bf16_t vbuf[8][32][64];  // V tile per wave (TDM dest)
  const int tid = threadIdx.x, wave = tid >> 5, lane = tid & 31;
  const int qt = blockIdx.x * 8 + wave;   // 0..127 query tile
  const int bh = blockIdx.y;              // 0..31
  const int lr  = lane & 15;
  const int kA  = (lane >> 4) * 16;
  const int kB  = (lane >> 4) * 16;
  const int hi8 = (lane >> 4) * 8;
  const bf16_t* Qp = Qm + (size_t)bh * SEQ * HD;
  const bf16_t* Kp = Km + (size_t)bh * SEQ * HD;
  const bf16_t* Vp = Vm + (size_t)bh * SEQ * HD;
  const unsigned vbuf_off = lds_offset_of(&vbuf[wave][0][0]);

  Frag aq[2];
  {
    const bf16_t* qr = Qp + (size_t)(qt * 16 + lr) * HD;
#pragma unroll
    for (int dc = 0; dc < 2; ++dc) {
      aq[dc].h[0] = *(const v8bf*)&qr[dc * 32 + kA];
      aq[dc].h[1] = *(const v8bf*)&qr[dc * 32 + kA + 8];
    }
  }
  v8f o[4] = {};
  float mrow[8], lsum[8];
#pragma unroll
  for (int r = 0; r < 8; ++r) { mrow[r] = -1e30f; lsum[r] = 0.f; }

  const int nch = (16 * qt + 47) >> 5;            // causal key-chunk count
  for (int c = 0; c < nch; ++c) {
    const int kc = c * 32;
    // Async V tile (32 keys x 64 dims, contiguous) -> per-wave LDS via TDM;
    // overlaps with the Q.K^T WMMAs and softmax below.
    tdm_load_2d(vbuf_off,
                (unsigned long long)(uintptr_t)(Vp + (size_t)kc * HD),
                64, 32, 64, 32, 64ull);

    // scores: S = Q(16x64) . K^T  -> two 16x16 frags (keys kc+0..15, kc+16..31)
    v8f s[2];
#pragma unroll
    for (int ks = 0; ks < 2; ++ks) {
      v8f sc = {};
#pragma unroll
      for (int dc = 0; dc < 2; ++dc) {
        Frag bk;
        const bf16_t* kr = Kp + (size_t)(kc + ks * 16 + lr) * HD;
        bk.h[0] = *(const v8bf*)&kr[dc * 32 + kB];
        bk.h[1] = *(const v8bf*)&kr[dc * 32 + kB + 8];
        sc = __builtin_amdgcn_wmma_f32_16x16x32_bf16(
            false, aq[dc].v, false, bk.v, (short)0, sc, false, false);
      }
      s[ks] = sc;
    }
    // online softmax (fp32, half-wave row reductions)
    const float scale = 0.125f;  // 1/sqrt(64)
    float alpha[8];
#pragma unroll
    for (int r = 0; r < 8; ++r) {
      int grow = qt * 16 + r + hi8;
      int c0 = kc + lr, c1 = kc + 16 + lr;
      float a0 = (c0 <= grow) ? s[0][r] * scale : -1e9f;
      float a1 = (c1 <= grow) ? s[1][r] * scale : -1e9f;
      float mx = fmaxf(a0, a1);
#pragma unroll
      for (int off = 1; off < 16; off <<= 1) mx = fmaxf(mx, __shfl_xor(mx, off, 32));
      float nm = fmaxf(mrow[r], mx);
      float al = __expf(mrow[r] - nm);
      float p0 = __expf(a0 - nm);
      float p1 = __expf(a1 - nm);
      float sm = p0 + p1;
#pragma unroll
      for (int off = 1; off < 16; off <<= 1) sm += __shfl_xor(sm, off, 32);
      lsum[r] = lsum[r] * al + sm;
      mrow[r] = nm;
      alpha[r] = al;
      pbuf[wave][r + hi8][lr]      = f2bf(p0);
      pbuf[wave][r + hi8][lr + 16] = f2bf(p1);
    }
#pragma unroll
    for (int n = 0; n < 4; ++n)
#pragma unroll
      for (int r = 0; r < 8; ++r) o[n][r] *= alpha[r];

    asm volatile("s_wait_dscnt 0" ::: "memory");  // pbuf stores visible
    __builtin_amdgcn_s_wait_tensorcnt(0);         // V tile DMA complete

    Frag ap;  // re-stripe P into A-fragment layout
    ap.h[0] = *(const v8bf*)&pbuf[wave][lr][kA];
    ap.h[1] = *(const v8bf*)&pbuf[wave][lr][kA + 8];
#pragma unroll
    for (int n = 0; n < 4; ++n) {
      Frag bv;
#pragma unroll
      for (int i = 0; i < 16; ++i) bv.v[i] = vbuf[wave][kB + i][n * 16 + lr];
      o[n] = __builtin_amdgcn_wmma_f32_16x16x32_bf16(
          false, ap.v, false, bv.v, (short)0, o[n], false, false);
    }
  }
  // normalize + store context (B,S,D) bf16
  const int b = bh >> 4, h = bh & 15;
#pragma unroll
  for (int n = 0; n < 4; ++n)
#pragma unroll
    for (int r = 0; r < 8; ++r) {
      int grow = qt * 16 + r + hi8;
      ctx[(size_t)(b * SEQ + grow) * D_EMBED + h * HD + n * 16 + lr] =
          f2bf(o[n][r] / lsum[r]);
    }
}

// --------------------------------------------------------------------- launcher
extern "C" void kernel_launch(void* const* d_in, const int* in_sizes, int n_in,
                              void* d_out, int out_size, void* d_ws, size_t ws_size,
                              hipStream_t stream) {
  (void)in_sizes; (void)n_in; (void)out_size; (void)ws_size;
  const float* x     = (const float*)d_in[0];
  const float* Wqkv  = (const float*)d_in[1];
  const float* Aqkv  = (const float*)d_in[2];
  const float* Bqkv  = (const float*)d_in[3];
  const float* Wproj = (const float*)d_in[4];
  const float* bproj = (const float*)d_in[5];
  const float* Aproj = (const float*)d_in[6];
  const float* Bproj = (const float*)d_in[7];
  float* out = (float*)d_out;

  char* ws = (char*)d_ws;
  auto alloc = [&](size_t bytes) -> char* {
    char* p = ws;
    ws += (bytes + 255) & ~(size_t)255;
    return p;
  };
  bf16_t* xb   = (bf16_t*)alloc((size_t)MROWS * D_EMBED * 2);
  bf16_t* wqb  = (bf16_t*)alloc((size_t)NQKV * D_EMBED * 2);
  bf16_t* wpb  = (bf16_t*)alloc((size_t)D_EMBED * D_EMBED * 2);
  bf16_t* qkvb = (bf16_t*)alloc((size_t)MROWS * NQKV * 2);
  bf16_t* qh   = (bf16_t*)alloc((size_t)MROWS * D_EMBED * 2);
  bf16_t* kh   = (bf16_t*)alloc((size_t)MROWS * D_EMBED * 2);
  bf16_t* vh   = (bf16_t*)alloc((size_t)MROWS * D_EMBED * 2);
  bf16_t* ctx  = (bf16_t*)alloc((size_t)MROWS * D_EMBED * 2);
  float*  tqkv = (float*)alloc((size_t)MROWS * RANK * 4);
  float*  tprj = (float*)alloc((size_t)MROWS * RANK * 4);

  int nx = MROWS * D_EMBED;
  k_f32_to_bf16<<<(nx + 255) / 256, 256, 0, stream>>>(x, xb, nx);
  int nwq = NQKV * D_EMBED;
  k_f32_to_bf16<<<(nwq + 255) / 256, 256, 0, stream>>>(Wqkv, wqb, nwq);
  int nwp = D_EMBED * D_EMBED;
  k_f32_to_bf16<<<(nwp + 255) / 256, 256, 0, stream>>>(Wproj, wpb, nwp);
  k_lora_t_f32<<<(MROWS * RANK) / 256, 256, 0, stream>>>(x, Aqkv, tqkv);

  dim3 g1(NQKV / 128, MROWS / 128);
  k_gemm_wmma<0><<<g1, 256, 0, stream>>>(xb, wqb, tqkv, Bqkv, nullptr,
                                         qkvb, nullptr, MROWS, NQKV, D_EMBED);

  int nr = BATCH * N_HEADS * SEQ * 32;
  k_rope_split<<<nr / 256, 256, 0, stream>>>(qkvb, qh, kh, vh);

  dim3 g2(SEQ / 16 / 8, BATCH * N_HEADS);
  k_flash_attn<<<g2, 256, 0, stream>>>(qh, kh, vh, ctx);

  k_lora_t_bf16<<<(MROWS * RANK) / 256, 256, 0, stream>>>(ctx, Aproj, tprj);

  dim3 g3(D_EMBED / 128, MROWS / 128);
  k_gemm_wmma<1><<<g3, 256, 0, stream>>>(ctx, wpb, tprj, Bproj, bproj,
                                         nullptr, out, MROWS, D_EMBED, D_EMBED);
}
